// ES_RNN_S_26671746908435
// MI455X (gfx1250) — matrix operand order repaired and checked by reading, not compiled, not run on hardware
//
#include <hip/hip_runtime.h>
#include <hip/hip_bf16.h>
#include <math.h>

// ---- problem constants (match reference setup) ----
#define T_LEN 50000
#define S_LEN 168
#define WIN   336
#define OUTW  48
#define NWIN  (T_LEN - WIN - OUTW + 1)   // 49617
#define KPAD  352                        // 336 padded to 11 tiles of 32
#define KT    11
#define NT1   21                         // 336/16 col tiles of W1
#define NT2   3                          // 48/16 col tiles of W2
#define GW    2                          // waves per gemm block

typedef __attribute__((ext_vector_type(16))) _Float16 v16h;
typedef __attribute__((ext_vector_type(8)))  _Float16 v8h;
typedef __attribute__((ext_vector_type(8)))  float    v8f;

// -------------------------------------------------------------------------
// 1) Sequential Holt-Winters scan: levels[T], seas[S+T], level_var_loss.
// -------------------------------------------------------------------------
__global__ void es_scan_kernel(const float* __restrict__ x,
                               const float* __restrict__ level_sc,
                               const float* __restrict__ seas_sc,
                               const float* __restrict__ init_seas,
                               const int*   __restrict__ lvp,
                               float* __restrict__ levels,
                               float* __restrict__ seas,
                               float* __restrict__ loss_out) {
    if (threadIdx.x != 0 || blockIdx.x != 0) return;
    const float a = 1.0f / (1.0f + __expf(-level_sc[0]));
    const float g = 1.0f / (1.0f + __expf(-seas_sc[0]));
    for (int s = 0; s < S_LEN; ++s) seas[s] = __expf(init_seas[s]);
    seas[S_LEN] = seas[0];
    float level = x[0] / seas[0];
    levels[0] = level;
    float loglv = __logf(level);
    float prev_ld = 0.0f;
    double acc = 0.0;
    for (int i = 1; i < T_LEN; ++i) {
        const float si = seas[i];
        const float xi = x[i];
        const float nl = a * xi / si + (1.0f - a) * level;
        seas[S_LEN + i] = g * xi / nl + (1.0f - g) * si;
        levels[i] = nl;
        const float lognl = __logf(nl);
        const float ld = lognl - loglv;
        if (i >= 2) { const float dd = ld - prev_ld; acc += (double)dd * (double)dd; }
        prev_ld = ld; loglv = lognl; level = nl;
    }
    loss_out[0] = (float)(acc / (double)(T_LEN - 2)) * (float)lvp[0];
}

// -------------------------------------------------------------------------
// 2) Avec[t] = log(x[t]/seas[t])
// -------------------------------------------------------------------------
__global__ void avec_kernel(const float* __restrict__ x,
                            const float* __restrict__ seas,
                            float* __restrict__ Avec) {
    int t = blockIdx.x * blockDim.x + threadIdx.x;
    if (t < T_LEN) Avec[t] = __logf(x[t] / seas[t]);
}

// -------------------------------------------------------------------------
// 3) loglev[n] = log(levels[n+WIN])
// -------------------------------------------------------------------------
__global__ void loglev_kernel(const float* __restrict__ levels,
                              float* __restrict__ loglev) {
    int n = blockIdx.x * blockDim.x + threadIdx.x;
    if (n < NWIN) loglev[n] = __logf(levels[n + WIN]);
}

// -------------------------------------------------------------------------
// 4) labels[n,o] = Avec[n+WIN+o] - loglev[n]
// -------------------------------------------------------------------------
__global__ void labels_kernel(const float* __restrict__ Avec,
                              const float* __restrict__ loglev,
                              float* __restrict__ labels) {
    int t = blockIdx.x * blockDim.x + threadIdx.x;
    if (t >= NWIN * OUTW) return;
    int n = t / OUTW;
    int o = t - n * OUTW;
    labels[t] = Avec[n + WIN + o] - loglev[n];
}

// -------------------------------------------------------------------------
// 5) Pack W1/W2 (f32 row-major) into f16 WMMA B-fragments:
//    frag (kt,nt), lane L: 16 halves = B[k, col] for
//      col = nt*16 + (L&15), k = kt*32 + ((L>>4)<<4) + j, j=0..15; k>=K -> 0
//    stored contiguously so the GEMM loads one aligned v16h per fragment.
// -------------------------------------------------------------------------
__global__ void pack_w1_kernel(const float* __restrict__ W1,
                               _Float16* __restrict__ w1p) {
    int t = blockIdx.x * blockDim.x + threadIdx.x;
    if (t >= KT * NT1 * 32) return;
    int lane = t & 31, f = t >> 5;
    int kt = f / NT1, nt = f % NT1;
    int col = nt * 16 + (lane & 15);
    int kb  = kt * 32 + ((lane >> 4) << 4);
    _Float16* dst = w1p + (size_t)t * 16;
    #pragma unroll
    for (int j = 0; j < 16; ++j) {
        int k = kb + j;
        float v = (k < WIN) ? W1[(size_t)k * WIN + col] : 0.0f;
        dst[j] = (_Float16)v;
    }
}

__global__ void pack_w2_kernel(const float* __restrict__ W2,
                               _Float16* __restrict__ w2p) {
    int t = blockIdx.x * blockDim.x + threadIdx.x;
    if (t >= KT * NT2 * 32) return;
    int lane = t & 31, f = t >> 5;
    int kt = f / NT2, nt = f % NT2;
    int col = nt * 16 + (lane & 15);
    int kb  = kt * 32 + ((lane >> 4) << 4);
    _Float16* dst = w2p + (size_t)t * 16;
    #pragma unroll
    for (int j = 0; j < 16; ++j) {
        int k = kb + j;
        float v = (k < WIN) ? W2[(size_t)k * OUTW + col] : 0.0f;
        dst[j] = (_Float16)v;
    }
}

// -------------------------------------------------------------------------
// 6) Fused GEMM: out = tanh(inp @ W1 + b1) @ W2 + b2.
//    One wave per 16-row slab; 2 waves / 64-thread block.
//    inp tile and h tile live in per-wave LDS (f16, K padded to 352).
// -------------------------------------------------------------------------
__global__ void __launch_bounds__(64)
gemm_kernel(const float* __restrict__ Avec,
            const float* __restrict__ loglev,
            const float* __restrict__ noise,
            const _Float16* __restrict__ w1p,
            const _Float16* __restrict__ w2p,
            const float* __restrict__ b1,
            const float* __restrict__ b2,
            float* __restrict__ out) {
    __shared__ _Float16 ibuf[GW][16 * KPAD];
    __shared__ _Float16 hbuf[GW][16 * KPAD];

    const int lane = threadIdx.x & 31;
    const int wv   = threadIdx.x >> 5;
    const int slab = blockIdx.x * GW + wv;
    const int n0   = slab * 16;
    if (n0 >= NWIN) return;                 // wave-uniform exit

    // ---- stage inp tile (16 x KPAD, f16) into LDS; noise read once ----
    _Float16* ib = ibuf[wv];
    for (int r = 0; r < 16; ++r) {
        int nr = n0 + r; if (nr > NWIN - 1) nr = NWIN - 1;  // clamp tail rows
        const float ll = loglev[nr];
        const float* av = Avec + nr;
        const float* nz = noise + (size_t)nr * WIN;
        for (int c0 = 0; c0 < KPAD; c0 += 32) {
            int c = c0 + lane;
            float v = 0.0f;
            if (c < WIN) v = av[c] - ll + nz[c];
            ib[r * KPAD + c] = (_Float16)v;
        }
    }
    // per-wave LDS ordering (DS in-order within wave) makes data visible.

    const int row   = lane & 15;            // A-fragment row M
    const int koff  = (lane >> 4) ? 16 : 0; // contiguous k order (as packed B): lanes>=16 take K run 16..31
    const int colq  = lane & 15;            // C/D layout column within tile
    const int mbase = (lane >> 4) << 3;     // C/D layout: lanes>=16 hold M=8..15
    const _Float16* arow = ib + row * KPAD;
    _Float16* hb = hbuf[wv];

    // ---- GEMM1: h = tanh(inp @ W1 + b1), stored f16 in LDS ----
    for (int nt = 0; nt < NT1; ++nt) {
        v8f acc = {};
        for (int kt = 0; kt < KT; ++kt) {
            const int kb = kt * 32 + koff;
            v8h lo = *(const v8h*)(arow + kb);       // K run kb..kb+7
            v8h hi = *(const v8h*)(arow + kb + 8);   // K run kb+8..kb+15
            v16h afrag = __builtin_shufflevector(lo, hi,
                0,1,2,3,4,5,6,7,8,9,10,11,12,13,14,15);
            v16h bfrag = *(const v16h*)(w1p + ((size_t)(kt * NT1 + nt) * 32 + lane) * 16);
            acc = __builtin_amdgcn_wmma_f32_16x16x32_f16(
                false, afrag, false, bfrag, (short)0, acc, false, false);
        }
        const int col = nt * 16 + colq;
        const float bias = b1[col];
        #pragma unroll
        for (int v = 0; v < 8; ++v) {
            hb[(mbase + v) * KPAD + col] = (_Float16)tanhf(acc[v] + bias);
        }
    }
    // zero the K-pad columns 336..351 of the h tile (16x16 halves, 8/lane)
    #pragma unroll
    for (int j = 0; j < 8; ++j) {
        int idx = lane + 32 * j;
        hb[(idx >> 4) * KPAD + WIN + (idx & 15)] = (_Float16)0.0f;
    }

    // ---- GEMM2: out = h @ W2 + b2 ----
    const _Float16* hrow = hb + row * KPAD;
    v8f acc2[NT2] = {};
    for (int kt = 0; kt < KT; ++kt) {
        const int kb = kt * 32 + koff;
        v8h lo = *(const v8h*)(hrow + kb);
        v8h hi = *(const v8h*)(hrow + kb + 8);
        v16h afrag = __builtin_shufflevector(lo, hi,
            0,1,2,3,4,5,6,7,8,9,10,11,12,13,14,15);
        #pragma unroll
        for (int nt = 0; nt < NT2; ++nt) {
            v16h bfrag = *(const v16h*)(w2p + ((size_t)(kt * NT2 + nt) * 32 + lane) * 16);
            acc2[nt] = __builtin_amdgcn_wmma_f32_16x16x32_f16(
                false, afrag, false, bfrag, (short)0, acc2[nt], false, false);
        }
    }
    #pragma unroll
    for (int nt = 0; nt < NT2; ++nt) {
        const int c = nt * 16 + colq;           // < 48
        const float bias = b2[c];
        #pragma unroll
        for (int v = 0; v < 8; ++v) {
            const int nn = n0 + mbase + v;
            if (nn < NWIN) out[(size_t)nn * OUTW + c] = acc2[nt][v] + bias;
        }
    }
}

// -------------------------------------------------------------------------
// Launch
// -------------------------------------------------------------------------
extern "C" void kernel_launch(void* const* d_in, const int* in_sizes, int n_in,
                              void* d_out, int out_size, void* d_ws, size_t ws_size,
                              hipStream_t stream) {
    const float* x         = (const float*)d_in[0];
    const float* noise     = (const float*)d_in[1];
    const float* level_sc  = (const float*)d_in[2];
    const float* seas_sc   = (const float*)d_in[3];
    const float* init_seas = (const float*)d_in[4];
    const float* W1        = (const float*)d_in[5];
    const float* b1        = (const float*)d_in[6];
    const float* W2        = (const float*)d_in[7];
    const float* b2        = (const float*)d_in[8];
    const int*   lvp       = (const int*)d_in[11];

    float* out    = (float*)d_out;                    // [NWIN*48]
    float* labels = out + (size_t)NWIN * OUTW;        // [NWIN*48]
    float* loss   = labels + (size_t)NWIN * OUTW;     // [1]

    // workspace layout
    char* ws = (char*)d_ws;
    float* levels = (float*)ws;                       // T
    float* seas   = levels + T_LEN;                   // S+T
    float* Avec   = seas + (S_LEN + T_LEN);           // T
    float* loglev = Avec + T_LEN;                     // NWIN
    size_t off = (size_t)(levels + T_LEN + (S_LEN + T_LEN) + T_LEN + NWIN - (float*)ws) * sizeof(float);
    off = (off + 31) & ~(size_t)31;
    _Float16* w1p = (_Float16*)(ws + off);            // KT*NT1*32*16 halves
    _Float16* w2p = w1p + (size_t)KT * NT1 * 32 * 16; // KT*NT2*32*16 halves

    es_scan_kernel<<<1, 32, 0, stream>>>(x, level_sc, seas_sc, init_seas, lvp,
                                         levels, seas, loss);
    avec_kernel<<<(T_LEN + 255) / 256, 256, 0, stream>>>(x, seas, Avec);
    loglev_kernel<<<(NWIN + 255) / 256, 256, 0, stream>>>(levels, loglev);
    labels_kernel<<<(NWIN * OUTW + 255) / 256, 256, 0, stream>>>(Avec, loglev, labels);
    pack_w1_kernel<<<(KT * NT1 * 32 + 255) / 256, 256, 0, stream>>>(W1, w1p);
    pack_w2_kernel<<<(KT * NT2 * 32 + 255) / 256, 256, 0, stream>>>(W2, w2p);

    const int slabs  = (NWIN + 15) / 16;              // 3102
    const int blocks = (slabs + GW - 1) / GW;         // 1551
    gemm_kernel<<<blocks, GW * 32, 0, stream>>>(Avec, loglev, noise,
                                                w1p, w2p, b1, b2, out);
}
